// NaiveSDPA_65841848648347
// MI455X (gfx1250) — matrix operand order repaired and checked by reading, not compiled, not run on hardware
//
#include <hip/hip_runtime.h>

typedef __attribute__((ext_vector_type(16))) _Float16 v16h;
typedef __attribute__((ext_vector_type(8)))  _Float16 v8h;
typedef __attribute__((ext_vector_type(4)))  _Float16 v4h;
typedef __attribute__((ext_vector_type(8)))  float    v8f;

constexpr int N_ = 4, H_ = 16, S_ = 2048, SKV_ = 2048, K_ = 128, V_ = 128;
constexpr float SCALE = 0.08838834764831845f;  // 128^-0.5 (folded into Q)
constexpr float NEGI  = -1e30f;

// LDS row strides (in f16 units), padded against bank conflicts
constexpr int KLD = 136;   // K tile  [32 kv][128 hd], row-major
constexpr int VLD = 40;    // V tile  [128 vcol][32 kv], transposed
constexpr int PLD = 40;    // per-wave P scratch [16 q][32 kv]

__device__ inline float rmax16(float x) {
  x = fmaxf(x, __shfl_xor(x, 1, 32));
  x = fmaxf(x, __shfl_xor(x, 2, 32));
  x = fmaxf(x, __shfl_xor(x, 4, 32));
  x = fmaxf(x, __shfl_xor(x, 8, 32));
  return x;
}
__device__ inline float rsum16(float x) {
  x += __shfl_xor(x, 1, 32);
  x += __shfl_xor(x, 2, 32);
  x += __shfl_xor(x, 4, 32);
  x += __shfl_xor(x, 8, 32);
  return x;
}

// build a 16-f16 fragment from two 8-f16 contiguous LDS runs (each 16B aligned)
__device__ inline v16h lds_frag(const _Float16* p0, const _Float16* p1) {
  v8h lo = *(const v8h*)p0;
  v8h hi = *(const v8h*)p1;
  return __builtin_shufflevector(lo, hi, 0,1,2,3,4,5,6,7,8,9,10,11,12,13,14,15);
}

__global__ __launch_bounds__(256)
void NaiveSDPA_flash_wmma_kernel(const float* __restrict__ q,
                                 const float* __restrict__ k,
                                 const float* __restrict__ v,
                                 const unsigned char* __restrict__ kpm,
                                 float* __restrict__ out)
{
  __shared__ _Float16 Klds[2][32 * KLD];       // 2 x  8704 B (double buffered)
  __shared__ _Float16 Vlds[2][128 * VLD];      // 2 x 10240 B (V transposed)
  __shared__ _Float16 Pws [8 * 16 * PLD];      //     10240 B (per-wave scratch)

  const int tid  = threadIdx.x;
  const int wave = tid >> 5;
  const int lane = tid & 31;
  const int lo   = lane & 15;
  const int half = lane >> 4;

  const int bh = blockIdx.y;            // b*H + h
  const int b  = bh / H_;
  const int qb = blockIdx.x * 128;      // block's first query row
  const int qw = qb + wave * 16;        // this wave's first query row

  const float* qp  = q + (size_t)bh * S_   * K_;
  const float* kp  = k + (size_t)bh * SKV_ * K_;
  const float* vp  = v + (size_t)bh * SKV_ * V_;
  const unsigned char* kpb = kpm + (size_t)b * SKV_;
  _Float16* Pw = Pws + wave * 16 * PLD;

  // cooperative-staging coordinates (same for every tile)
  const int sr = tid >> 5;          // rows 0..7 (+8 per pass)   [tid*4 / 128]
  const int sc = (tid & 31) * 4;    // col 0..124 step 4

  // ---- Q tile -> 4 A-fragments, pre-scaled by 128^-0.5, f32 -> f16 ----
  const float* qbase = qp + (size_t)(qw + lo) * K_;
  v16h qf[4];
#pragma unroll
  for (int kc = 0; kc < 4; ++kc) {
    const float4* p0 = (const float4*)(qbase + kc*32 + half*8);       // elems 0-7
    const float4* p1 = (const float4*)(qbase + kc*32 + 16 + half*8);  // elems 8-15
    float4 a0 = p0[0], a1 = p0[1], b0 = p1[0], b1 = p1[1];
    v16h f;
    f[0] =(_Float16)(a0.x*SCALE); f[1] =(_Float16)(a0.y*SCALE);
    f[2] =(_Float16)(a0.z*SCALE); f[3] =(_Float16)(a0.w*SCALE);
    f[4] =(_Float16)(a1.x*SCALE); f[5] =(_Float16)(a1.y*SCALE);
    f[6] =(_Float16)(a1.z*SCALE); f[7] =(_Float16)(a1.w*SCALE);
    f[8] =(_Float16)(b0.x*SCALE); f[9] =(_Float16)(b0.y*SCALE);
    f[10]=(_Float16)(b0.z*SCALE); f[11]=(_Float16)(b0.w*SCALE);
    f[12]=(_Float16)(b1.x*SCALE); f[13]=(_Float16)(b1.y*SCALE);
    f[14]=(_Float16)(b1.z*SCALE); f[15]=(_Float16)(b1.w*SCALE);
    qf[kc] = f;
  }

  v8f oacc[8];
  float mrun[8], lrun[8];
#pragma unroll
  for (int t = 0; t < 8; ++t)
#pragma unroll
    for (int j = 0; j < 8; ++j) oacc[t][j] = 0.0f;
#pragma unroll
  for (int r = 0; r < 8; ++r) { mrun[r] = NEGI; lrun[r] = 0.0f; }

  const int kvEnd = qb + 128;   // block-uniform causal bound (<= SKV_)

  // ---- prologue: stage tile 0 into buffer 0 ----
  {
#pragma unroll
    for (int pass = 0; pass < 4; ++pass) {
      const int r = sr + pass * 8;
      float4 kx = *(const float4*)(kp + (size_t)r * K_ + sc);
      float4 vx = *(const float4*)(vp + (size_t)r * V_ + sc);
      v4h kh; kh[0]=(_Float16)kx.x; kh[1]=(_Float16)kx.y;
              kh[2]=(_Float16)kx.z; kh[3]=(_Float16)kx.w;
      *(v4h*)(&Klds[0][r * KLD + sc]) = kh;
      Vlds[0][(sc + 0) * VLD + r] = (_Float16)vx.x;
      Vlds[0][(sc + 1) * VLD + r] = (_Float16)vx.y;
      Vlds[0][(sc + 2) * VLD + r] = (_Float16)vx.z;
      Vlds[0][(sc + 3) * VLD + r] = (_Float16)vx.w;
    }
  }
  __syncthreads();

  int buf = 0;
  for (int kv0 = 0; kv0 < kvEnd; kv0 += 32, buf ^= 1) {
    const int kvn = kv0 + 32;
    const bool have_next = (kvn < kvEnd);

    // ---- issue next tile's global loads NOW (latency hidden by compute) ----
    float4 kx[4], vx[4];
    if (have_next) {
#pragma unroll
      for (int pass = 0; pass < 4; ++pass) {
        const int r = sr + pass * 8;
        kx[pass] = *(const float4*)(kp + (size_t)(kvn + r) * K_ + sc);
        vx[pass] = *(const float4*)(vp + (size_t)(kvn + r) * V_ + sc);
      }
      if (kvn + 32 < kvEnd) {   // warm L2 two tiles ahead
        __builtin_prefetch(kp + (size_t)(kvn + 32) * K_ + tid * 16, 0, 1);
        __builtin_prefetch(vp + (size_t)(kvn + 32) * V_ + tid * 16, 0, 1);
      }
    }

    if (kv0 < qw + 16) {   // wave-uniform: tile intersects this wave's causal region
      const _Float16* Kb = Klds[buf];
      const _Float16* Vb = Vlds[buf];

      // ---- S = Q * K^T : two 16x16 tiles, K-dim 128 = 4 x 32 ----
      v8f c0, c1;
#pragma unroll
      for (int j = 0; j < 8; ++j) { c0[j] = 0.0f; c1[j] = 0.0f; }
#pragma unroll
      for (int kc = 0; kc < 4; ++kc) {
        v16h kb0 = lds_frag(&Kb[(     lo) * KLD + kc*32 + half*8],
                            &Kb[(     lo) * KLD + kc*32 + 16 + half*8]);
        c0 = __builtin_amdgcn_wmma_f32_16x16x32_f16(false, qf[kc], false, kb0,
                                                    (short)0, c0, false, false);
        v16h kb1 = lds_frag(&Kb[(16 + lo) * KLD + kc*32 + half*8],
                            &Kb[(16 + lo) * KLD + kc*32 + 16 + half*8]);
        c1 = __builtin_amdgcn_wmma_f32_16x16x32_f16(false, qf[kc], false, kb1,
                                                    (short)0, c1, false, false);
      }

      // ---- masked online softmax (rows owned per (half, reg)) ----
      const int col0 = kv0 + lo;
      const int col1 = kv0 + 16 + lo;
      const bool kpv0 = kpb[col0] != 0;
      const bool kpv1 = kpb[col1] != 0;
#pragma unroll
      for (int r = 0; r < 8; ++r) {
        const int row = qw + 8*half + r;
        const bool v0 = kpv0 && (col0 <= row);
        const bool v1 = kpv1 && (col1 <= row);
        float s0 = v0 ? c0[r] : NEGI;      // scale already folded into Q
        float s1 = v1 ? c1[r] : NEGI;
        float mt = rmax16(fmaxf(s0, s1));
        float mn = fmaxf(mrun[r], mt);
        float so = __expf(mrun[r] - mn);          // rescale of old state
        float p0 = v0 ? __expf(s0 - mn) : 0.0f;   // explicit 0 for masked cols
        float p1 = v1 ? __expf(s1 - mn) : 0.0f;
        float st = rsum16(p0 + p1);
        lrun[r] = lrun[r] * so + st;
        mrun[r] = mn;
#pragma unroll
        for (int tv = 0; tv < 8; ++tv) oacc[tv][r] *= so;
        Pw[(8*half + r) * PLD + lo]      = (_Float16)p0;
        Pw[(8*half + r) * PLD + 16 + lo] = (_Float16)p1;
      }

      // ---- O += P * V : 8 WMMAs over the 128 v-columns ----
      v16h pf = lds_frag(&Pw[lo * PLD + half*8],
                         &Pw[lo * PLD + 16 + half*8]);
#pragma unroll
      for (int tv = 0; tv < 8; ++tv) {
        v16h vb = lds_frag(&Vb[(tv*16 + lo) * VLD + half*8],
                           &Vb[(tv*16 + lo) * VLD + 16 + half*8]);
        oacc[tv] = __builtin_amdgcn_wmma_f32_16x16x32_f16(false, pf, false, vb,
                                                          (short)0, oacc[tv], false, false);
      }
    }

    // ---- convert + store next tile into the alternate buffer ----
    if (have_next) {
      _Float16* Kn = Klds[buf ^ 1];
      _Float16* Vn = Vlds[buf ^ 1];
#pragma unroll
      for (int pass = 0; pass < 4; ++pass) {
        const int r = sr + pass * 8;
        v4h kh; kh[0]=(_Float16)kx[pass].x; kh[1]=(_Float16)kx[pass].y;
                kh[2]=(_Float16)kx[pass].z; kh[3]=(_Float16)kx[pass].w;
        *(v4h*)(&Kn[r * KLD + sc]) = kh;
        Vn[(sc + 0) * VLD + r] = (_Float16)vx[pass].x;
        Vn[(sc + 1) * VLD + r] = (_Float16)vx[pass].y;
        Vn[(sc + 2) * VLD + r] = (_Float16)vx[pass].z;
        Vn[(sc + 3) * VLD + r] = (_Float16)vx[pass].w;
      }
    }
    __syncthreads();   // one barrier per tile: publishes next buffer,
                       // and orders this tile's reads before its buffer is reused
  }

  // ---- epilogue: normalize and store ----
#pragma unroll
  for (int r = 0; r < 8; ++r) {
    const int row = qw + 8*half + r;
    const float inv = 1.0f / lrun[r];   // kpm[:,0]==True + causal => lrun > 0
    float* op = out + ((size_t)bh * S_ + row) * V_;
#pragma unroll
    for (int tv = 0; tv < 8; ++tv)
      op[tv*16 + lo] = oacc[tv][r] * inv;
  }
}

extern "C" void kernel_launch(void* const* d_in, const int* in_sizes, int n_in,
                              void* d_out, int out_size, void* d_ws, size_t ws_size,
                              hipStream_t stream) {
  (void)in_sizes; (void)n_in; (void)out_size; (void)d_ws; (void)ws_size;
  const float* seqs   = (const float*)d_in[0];
  const float* keys   = (const float*)d_in[1];
  const float* values = (const float*)d_in[2];
  const unsigned char* kpm = (const unsigned char*)d_in[3];  // jnp bool = 1 byte
  // d_in[4] (attn_mask) intentionally unused: causal -1e9 mask recomputed analytically
  float* out = (float*)d_out;

  dim3 grid(S_ / 128, N_ * H_);   // (16, 64)
  dim3 block(256);                // 8 wave32 per block
  NaiveSDPA_flash_wmma_kernel<<<grid, block, 0, stream>>>(seqs, keys, values, kpm, out);
}
